// LIFLayer_15633680958053
// MI455X (gfx1250) — compile-verified, matrix operand order checked by reading, not run on hardware
//
#include <hip/hip_runtime.h>
#include <math.h>

// ---------------- problem constants ----------------
#define Tn   512
#define Bn   64
#define Cc   512        // C_IN == C_OUT
#define NWG  16         // persistent workgroups, 32 output cols each
#define NT   (Cc / 16)  // 32 column tiles total

// LDS layout (single static array, base offset 0)
#define LDS_XA    0u        // 64 KB: x_t (phase1) / a_t (phase2), bf16[64][512]
#define LDS_SLOW  65536u    // 64 KB: slow_prev bf16[64][512]
#define LDS_WCTRL 131072u   // 64 KB: gate weights  [32 kt][2 ct][32 lane][16 h]
#define LDS_WMAT  196608u   // 32 KB: W             [16 kt][2 ct][32 lane][16 h]
#define LDS_TOTAL 229376

typedef __attribute__((ext_vector_type(16))) __bf16 v16bf;
typedef __attribute__((ext_vector_type(8)))  float  v8f;

struct BfPack { uint4 a, b; };   // 32 bytes == v16bf

// A operand (16x32 bf16, M x K): lane m = lane&15; halves 0..7 -> K k0..k0+7,
// halves 8..15 -> K k0+16..k0+23 (k0 = 8*(lane>>4)): two 16B loads 32B apart.
__device__ inline v16bf ld_a(const __bf16* p) {
    BfPack t;
    t.a = *(const uint4*)(p);
    t.b = *(const uint4*)(p + 16);
    return __builtin_bit_cast(v16bf, t);
}
// B operand: packed contiguous 32 bytes per lane.
__device__ inline v16bf ld_b(const __bf16* p) {
    BfPack t;
    t.a = *(const uint4*)(p);
    t.b = *(const uint4*)(p + 8);
    return __builtin_bit_cast(v16bf, t);
}

// CDNA5 async global->LDS copy (ASYNCcnt path).
__device__ inline void async_b128(unsigned lds_off, unsigned long long gaddr) {
    asm volatile("global_load_async_to_lds_b128 %0, %1, off"
                 :: "v"(lds_off), "v"(gaddr) : "memory");
}
__device__ inline void wait_async0()  { asm volatile("s_wait_asynccnt 0x0"  ::: "memory"); }
__device__ inline void wait_async16() { asm volatile("s_wait_asynccnt 0x10" ::: "memory"); }

// ---------------- init kernels ----------------
__global__ void zero_state(unsigned* bar, __bf16* slow_g) {
    int i = blockIdx.x * blockDim.x + threadIdx.x;
    if (i == 0) *bar = 0u;
    if (i < 2 * Bn * Cc) slow_g[i] = (__bf16)0.0f;   // both parity buffers
}

// Pack f32 weights into WMMA-B-operand order:
// dst[(((kt*NT + nt)*32 + lane)*16 + h)] holds src element for
//   n = nt*16 + (lane&15),  k = kt*32 + kmap(h, lane>>4)
__global__ void pack_weights(const float* __restrict__ src, __bf16* __restrict__ dst,
                             int KT, int trans, int srcld) {
    int idx = blockIdx.x * blockDim.x + threadIdx.x;
    int total = KT * NT * 32 * 16;
    if (idx >= total) return;
    int h    = idx & 15;
    int lane = (idx >> 4) & 31;
    int nt   = (idx >> 9) & 31;
    int kt   = idx >> 14;
    int hi   = lane >> 4;
    int n    = nt * 16 + (lane & 15);
    int kl   = (h < 8) ? (hi * 8 + h) : (16 + hi * 8 + (h - 8));
    int k    = kt * 32 + kl;
    float v  = trans ? src[n * srcld + k] : src[k * srcld + n];
    dst[idx] = (__bf16)v;
}

// ---------------- persistent recurrence kernel ----------------
__global__ __launch_bounds__(256) void lif_kernel(
    const float*  __restrict__ seq,     // [T][B][C] f32
    const float*  __restrict__ ctrl_b,  // [C] f32
    const __bf16* __restrict__ wctrl,   // packed gate weights [32][NT][32][16]
    const __bf16* __restrict__ wmat,    // packed W            [16][NT][32][16]
    __bf16* __restrict__ slow_g,        // double buffer [2][B*C] bf16
    __bf16* __restrict__ a_g,           // double buffer [2][B*C] bf16
    unsigned* __restrict__ bar,
    float* __restrict__ out)            // [B][C] f32
{
    __shared__ __align__(16) unsigned char smem[LDS_TOTAL];
    __bf16* ldsx = (__bf16*)(smem + LDS_XA);
    __bf16* ldss = (__bf16*)(smem + LDS_SLOW);
    __bf16* ldsw = (__bf16*)(smem + LDS_WCTRL);
    __bf16* ldsm = (__bf16*)(smem + LDS_WMAT);

    const int tid  = threadIdx.x;
    const int wg   = blockIdx.x;
    const int wave = tid >> 5, lane = tid & 31;
    const int rt = wave >> 1, ct = wave & 1;         // row-tile 0..3, col-tile 0..1
    const int hi = lane >> 4, nl = lane & 15;
    const int ntile = wg * 2 + ct;                   // global 16-col tile id
    const int col   = ntile * 16 + nl;               // this lane's output column
    const int arow  = rt * 16 + nl;                  // A-operand row for this lane
    const int kbase = hi * 8;

    float vst[8], fst[8], slw[8], ssum[8];
    int mrow[8];
#pragma unroll
    for (int i = 0; i < 8; ++i) {
        vst[i] = fst[i] = slw[i] = ssum[i] = 0.0f;
        mrow[i] = rt * 16 + i + 8 * hi;              // D-matrix row of element i
    }
    const float bg = ctrl_b[col];
    const float ln_aslow = -0.005012541823544286f;   // ln(0.995)

    // ---- one-time staging: weights (reused 512x) + slow[0], all async ----
    for (int j = tid; j < 4096; j += 256) {          // gate weights: 64 KB
        int kt = j >> 7, rem = j & 127, c2 = rem >> 6, r2 = rem & 63;
        const __bf16* src = wctrl + ((size_t)(kt * NT + wg * 2 + c2) * 512) + r2 * 8;
        async_b128(LDS_WCTRL + (unsigned)j * 16u, (unsigned long long)src);
    }
    for (int j = tid; j < 2048; j += 256) {          // W: 32 KB
        int kt = j >> 7, rem = j & 127, c2 = rem >> 6, r2 = rem & 63;
        const __bf16* src = wmat + ((size_t)(kt * NT + wg * 2 + c2) * 512) + r2 * 8;
        async_b128(LDS_WMAT + (unsigned)j * 16u, (unsigned long long)src);
    }
    for (int j = tid; j < 4096; j += 256)            // slow[parity 0] (zeros)
        async_b128(LDS_SLOW + (unsigned)j * 16u,
                   (unsigned long long)(slow_g + j * 8));
    wait_async0();
    __syncthreads();

    for (int t = 0; t < Tn; ++t) {
        const int p0 = t & 1, p1 = p0 ^ 1;

        // ---- stage x_t into LDS (f32 -> bf16); slow_prev already prefetched ----
        const float* xt = seq + (size_t)t * Bn * Cc;
        if (t + 1 < Tn)
            __builtin_prefetch(seq + (size_t)(t + 1) * Bn * Cc + tid * 128, 0, 1);
        for (int j = tid; j < Bn * Cc / 4; j += 256) {
            float4 f = ((const float4*)xt)[j];
            __bf16* d = ldsx + j * 4;
            d[0] = (__bf16)f.x; d[1] = (__bf16)f.y;
            d[2] = (__bf16)f.z; d[3] = (__bf16)f.w;
        }
        wait_async0();              // slow_prev prefetch (issued last step) landed
        __syncthreads();

        // ---- phase 1: gate GEMM [64x1024]x[1024x32], all operands in LDS ----
        v8f g = {};
        for (int kt = 0; kt < 32; ++kt) {
            const __bf16* abase = (kt < 16) ? ldsx : ldss;
            v16bf A = ld_a(abase + arow * Cc + (kt & 15) * 32 + kbase);
            v16bf B = ld_b(ldsw + ((kt * 2 + ct) * 32 + lane) * 16);
            g = __builtin_amdgcn_wmma_f32_16x16x32_bf16(
                    false, A, false, B, (short)0, g, false, false);
        }

        // ---- state update on this wave's 16x16 tile, publish slow/a (bf16) ----
        __bf16* snew = slow_g + (size_t)p1 * Bn * Cc;
        __bf16* anew = a_g    + (size_t)p0 * Bn * Cc;
#pragma unroll
        for (int i = 0; i < 8; ++i) {
            const int m = mrow[i];
            float x   = (float)ldsx[m * Cc + col];
            float z   = g[i] + bg;
            float sig = 1.0f / (1.0f + __expf(-z));
            float wrp = sig * 0.9f + 0.05f;
            float sl  = __expf(wrp * ln_aslow) * slw[i] + x;  slw[i] = sl;
            float f   = 0.9f * fst[i] + x;                    fst[i] = f;
            float a   = x + 0.5f * f + 0.5f * sl;
            snew[m * Cc + col] = (__bf16)sl;
            anew[m * Cc + col] = (__bf16)a;
        }

        // ---- device-scope release -> grid barrier -> acquire ----
        __builtin_amdgcn_fence(__ATOMIC_RELEASE, "agent");
        __syncthreads();
        if (tid == 0) {
            __hip_atomic_fetch_add(bar, 1u, __ATOMIC_RELEASE, __HIP_MEMORY_SCOPE_AGENT);
            while (__hip_atomic_load(bar, __ATOMIC_ACQUIRE, __HIP_MEMORY_SCOPE_AGENT)
                   < (unsigned)((t + 1) * NWG))
                __builtin_amdgcn_s_sleep(1);
        }
        __syncthreads();
        __builtin_amdgcn_fence(__ATOMIC_ACQUIRE, "agent");

        // ---- phase 2: async a_t -> ldsx, then prefetch slow[p1] -> ldss ----
        // (async loads retire in order: waiting to <=16 outstanding guarantees
        //  the a-copy landed while the slow prefetch overlaps the cur GEMM)
        const __bf16* afull = a_g + (size_t)p0 * Bn * Cc;
        const __bf16* snext = slow_g + (size_t)p1 * Bn * Cc;
        for (int j = tid; j < 4096; j += 256)
            async_b128(LDS_XA + (unsigned)j * 16u,
                       (unsigned long long)(afull + j * 8));
        for (int j = tid; j < 4096; j += 256)
            async_b128(LDS_SLOW + (unsigned)j * 16u,
                       (unsigned long long)(snext + j * 8));
        wait_async16();
        __syncthreads();

        // ---- cur GEMM [64x512]x[512x32], operands in LDS ----
        v8f c = {};
        for (int kt = 0; kt < 16; ++kt) {
            v16bf A = ld_a(ldsx + arow * Cc + kt * 32 + kbase);
            v16bf B = ld_b(ldsm + ((kt * 2 + ct) * 32 + lane) * 16);
            c = __builtin_amdgcn_wmma_f32_16x16x32_bf16(
                    false, A, false, B, (short)0, c, false, false);
        }
#pragma unroll
        for (int i = 0; i < 8; ++i) {
            float v = 0.9f * vst[i] + 0.1f * c[i];
            float s = (v - 1.0f > 0.0f) ? 1.0f : 0.0f;
            ssum[i] += s;
            vst[i] = (s > 0.0f) ? 0.0f : v;
        }
        __syncthreads();   // protect LDS reuse before next step's staging
    }

#pragma unroll
    for (int i = 0; i < 8; ++i)
        out[mrow[i] * Cc + col] = ssum[i] * (1.0f / (float)Tn);
}

// ---------------- host launch ----------------
extern "C" void kernel_launch(void* const* d_in, const int* in_sizes, int n_in,
                              void* d_out, int out_size, void* d_ws, size_t ws_size,
                              hipStream_t stream) {
    (void)in_sizes; (void)n_in; (void)out_size; (void)ws_size;
    const float* seq    = (const float*)d_in[0];   // [T*B*C]
    const float* W      = (const float*)d_in[1];   // [C*C]
    const float* ctrl_w = (const float*)d_in[2];   // [C * 2C]
    const float* ctrl_b = (const float*)d_in[3];   // [C]

    char* ws = (char*)d_ws;
    unsigned* bar   = (unsigned*)ws;                              // 256 B slot
    __bf16* slow_g  = (__bf16*)(ws + 256);                        // 128 KB
    __bf16* a_g     = (__bf16*)(ws + 256 + 131072);               // 128 KB
    __bf16* wctrl   = (__bf16*)(ws + 256 + 262144);               // 1 MB
    __bf16* wmat    = (__bf16*)(ws + 256 + 262144 + 1048576);     // 512 KB

    zero_state<<<(2 * Bn * Cc + 255) / 256, 256, 0, stream>>>(bar, slow_g);
    pack_weights<<<(32 * NT * 32 * 16 + 255) / 256, 256, 0, stream>>>(
        ctrl_w, wctrl, 32, 1, 2 * Cc);
    pack_weights<<<(16 * NT * 32 * 16 + 255) / 256, 256, 0, stream>>>(
        W, wmat, 16, 0, Cc);
    lif_kernel<<<NWG, 256, 0, stream>>>(seq, ctrl_b, wctrl, wmat,
                                        slow_g, a_g, bar, (float*)d_out);
}
